// MambaPredictor_79534204387654
// MI455X (gfx1250) — matrix operand order repaired and checked by reading, not compiled, not run on hardware
//
#include <hip/hip_runtime.h>
#include <cmath>

// Model dims (from reference)
#define BB      8
#define LL      4096
#define DIMM    384
#define DEPTHN  6
#define DSTATE  16
#define DCONVN  4
#define DINNER  768
#define DTRANK  24
#define BLROWS  (BB*LL)          // 32768 token rows

typedef __attribute__((ext_vector_type(16))) __bf16 v16bf;
typedef __attribute__((ext_vector_type(8)))  float  v8f;

// f32 -> bf16 pair pack: round-half-up then v_perm_b32 grabs both high halves.
// 3 VALU per 2 elements (vs ~5 for exact RNE bit twiddling).
__device__ __forceinline__ unsigned pack2bf(float lo, float hi) {
    unsigned a = __float_as_uint(lo) + 0x8000u;
    unsigned b = __float_as_uint(hi) + 0x8000u;
    // result = { b[31:16], a[31:16] } ; sel bytes 4..7 = src0(b), 0..3 = src1(a)
    return __builtin_amdgcn_perm(b, a, 0x07060302u);
}
__device__ __forceinline__ uint4 pack8bf(float4 a, float4 b) {
    uint4 o;
    o.x = pack2bf(a.x, a.y); o.y = pack2bf(a.z, a.w);
    o.z = pack2bf(b.x, b.y); o.w = pack2bf(b.z, b.w);
    return o;
}

union FragBF { uint4 q[2]; v16bf v; };

enum { EPI_NONE = 0, EPI_BIAS_SOFTPLUS = 1, EPI_BIAS_GELU = 2,
       EPI_RESADD = 3, EPI_BIAS_RESADD = 4 };

// LDS row stride in bf16 elements: 64 data + 8 pad = 144B rows (16B aligned slots)
#define LDK 72

// ---------------------------------------------------------------------------
// Tiled bf16 WMMA GEMM:  C[M,N] = A[M,K] * W[N,K]^T  (+ fused epilogue)
// Block tile 128(M) x 64(N), K-step 64 (2 WMMA sub-steps); 8 waves ->
// each wave a 32x32 C tile => 8 v_wmma per barrier pair.
// ALIGNED=true  : K%64==0 && N%64==0, zero guards anywhere.
// ALIGNED=false : guarded K/N tails (only the tiny xproj/dtproj GEMMs).
// ---------------------------------------------------------------------------
template<bool ALIGNED>
__global__ __launch_bounds__(256)
void gemm_wmma(const float* __restrict__ A, int lda,
               const float* __restrict__ W, int ldw,
               int K, int N,
               const float* __restrict__ bias,
               float* __restrict__ C, int ldc,
               const float* __restrict__ resid,
               int epi)
{
    __shared__ unsigned short As[128 * LDK];
    __shared__ unsigned short Bs[64 * LDK];

    const int tid  = threadIdx.x;
    const int lane = tid & 31;
    const int wv   = tid >> 5;      // 0..7
    const int wm   = wv & 3;        // wave M subtile (4)
    const int wn   = wv >> 2;       // wave N subtile (2)
    const int half = lane >> 4;     // lane half
    const int l16  = lane & 15;

    const int mBase = blockIdx.y * 128;
    const int nBase = blockIdx.x * 64;

    v8f acc[2][2] = {};

    // fill mapping: A: 128 rows x 64 cols -> thread t loads row t>>1, 32 cols
    const int ar = tid >> 1;
    const int ac = (tid & 1) * 32;
    // B: 64 rows x 64 cols -> thread t loads row t>>2, 16 cols
    const int br = tid >> 2;
    const int bc = (tid & 3) * 16;

    for (int k0 = 0; k0 < K; k0 += 64) {
        // ---------------- stage A tile (128 x 64) ----------------
        {
            const size_t gbase = (size_t)(mBase + ar) * lda + k0 + ac;
            unsigned short* dst = &As[ar * LDK + ac];
            if (ALIGNED || (k0 + ac + 32 <= K)) {
                // fast path: 8x global_load_b128 -> 4x ds_store_b128
                const float4* g = (const float4*)(A + gbase);
                #pragma unroll
                for (int i = 0; i < 4; ++i)
                    *(uint4*)(dst + 8 * i) = pack8bf(g[2 * i], g[2 * i + 1]);
                if (k0 + 64 < K)                   // gfx1250 global_prefetch_b8
                    __builtin_prefetch(&A[gbase + 64], 0, 0);
            } else {
                // K tail: guarded scalar loads, packed stores
                #pragma unroll
                for (int j = 0; j < 32; j += 2) {
                    const int kk = k0 + ac + j;
                    float v0 = (kk     < K) ? A[gbase + j]     : 0.0f;
                    float v1 = (kk + 1 < K) ? A[gbase + j + 1] : 0.0f;
                    *(unsigned*)(dst + j) = pack2bf(v0, v1);
                }
            }
        }
        // ---------------- stage W tile (64 rows x 64 K) ----------------
        {
            const int n = nBase + br;
            const size_t gbase = (size_t)n * ldw + k0 + bc;
            unsigned short* dst = &Bs[br * LDK + bc];
            if (ALIGNED || (n < N && k0 + bc + 16 <= K)) {
                const float4* g = (const float4*)(W + gbase);
                #pragma unroll
                for (int i = 0; i < 2; ++i)
                    *(uint4*)(dst + 8 * i) = pack8bf(g[2 * i], g[2 * i + 1]);
            } else if (n < N) {
                #pragma unroll
                for (int j = 0; j < 16; j += 2) {
                    const int kk = k0 + bc + j;
                    float v0 = (kk     < K) ? W[gbase + j]     : 0.0f;
                    float v1 = (kk + 1 < K) ? W[gbase + j + 1] : 0.0f;
                    *(unsigned*)(dst + j) = pack2bf(v0, v1);
                }
            } else {
                const uint4 z = {0u, 0u, 0u, 0u};
                *(uint4*)(dst)     = z;
                *(uint4*)(dst + 8) = z;
            }
        }
        __syncthreads();

        // ---------------- 2 x (fragment load + 4 WMMA) ----------------
        #pragma unroll
        for (int s = 0; s < 2; ++s) {
            const int ks = s * 32;  // bf16 col offset within tile
            FragBF a0, a1, b0, b1;
            {
                // A 16x32 bf16: lane<16 -> K 0..7 & 16..23 ; lane>=16 -> 8..15 & 24..31
                const int r0 = wm * 32 + l16;
                const unsigned short* p0 = &As[r0 * LDK + ks + half * 8];
                a0.q[0] = *(const uint4*)p0;
                a0.q[1] = *(const uint4*)(p0 + 16);
                const unsigned short* p1 = &As[(r0 + 16) * LDK + ks + half * 8];
                a1.q[0] = *(const uint4*)p1;
                a1.q[1] = *(const uint4*)(p1 + 16);
            }
            {
                // B 32x16 bf16: lane<16 -> col l16, K 0..7 & 16..23 ; lane>=16 -> K 8..15 & 24..31
                const int n0 = wn * 32 + l16;
                const unsigned short* p0 = &Bs[n0 * LDK + ks + half * 8];
                b0.q[0] = *(const uint4*)p0;
                b0.q[1] = *(const uint4*)(p0 + 16);
                const unsigned short* p1 = &Bs[(n0 + 16) * LDK + ks + half * 8];
                b1.q[0] = *(const uint4*)p1;
                b1.q[1] = *(const uint4*)(p1 + 16);
            }
            acc[0][0] = __builtin_amdgcn_wmma_f32_16x16x32_bf16(false, a0.v, false, b0.v,
                                                                (short)0, acc[0][0], false, false);
            acc[0][1] = __builtin_amdgcn_wmma_f32_16x16x32_bf16(false, a0.v, false, b1.v,
                                                                (short)0, acc[0][1], false, false);
            acc[1][0] = __builtin_amdgcn_wmma_f32_16x16x32_bf16(false, a1.v, false, b0.v,
                                                                (short)0, acc[1][0], false, false);
            acc[1][1] = __builtin_amdgcn_wmma_f32_16x16x32_bf16(false, a1.v, false, b1.v,
                                                                (short)0, acc[1][1], false, false);
        }
        __syncthreads();
    }

    // --- epilogue: C/D layout is row = 16i + r + 8*half, col = 16j + l16 ---
    #pragma unroll
    for (int i = 0; i < 2; ++i) {
        #pragma unroll
        for (int j = 0; j < 2; ++j) {
            #pragma unroll
            for (int r = 0; r < 8; ++r) {
                const int gRow = mBase + wm * 32 + i * 16 + r + half * 8;
                const int gCol = nBase + wn * 32 + j * 16 + l16;
                if (!ALIGNED && gCol >= N) continue;
                float v = acc[i][j][r];
                const size_t idx = (size_t)gRow * ldc + gCol;
                if (epi == EPI_BIAS_SOFTPLUS) {
                    v += bias[gCol];
                    v = (v > 20.0f) ? v : log1pf(expf(v));
                } else if (epi == EPI_BIAS_GELU) {
                    v += bias[gCol];
                    v = 0.5f * v * (1.0f + erff(v * 0.70710678118654752f));
                } else if (epi == EPI_RESADD) {
                    v += resid[idx];
                } else if (epi == EPI_BIAS_RESADD) {
                    v += bias[gCol] + resid[idx];
                }
                C[idx] = v;
            }
        }
    }
}

// ---------------------------------------------------------------------------
// x + pos_emb (pos broadcast over batch)
// ---------------------------------------------------------------------------
__global__ __launch_bounds__(256)
void add_pos(const float* __restrict__ x, const float* __restrict__ pos,
             float* __restrict__ out, int n, int per_batch)
{
    int i = blockIdx.x * 256 + threadIdx.x;
    if (i < n) out[i] = x[i] + pos[i % per_batch];
}

// ---------------------------------------------------------------------------
// LayerNorm: one wave32 per row (D = 384), shuffle reduction
// ---------------------------------------------------------------------------
__global__ __launch_bounds__(256)
void ln_rows(const float* __restrict__ x, const float* __restrict__ sc,
             const float* __restrict__ bi, float* __restrict__ y,
             int rows, int D)
{
    const int wid = threadIdx.x >> 5, lane = threadIdx.x & 31;
    const int row = blockIdx.x * 8 + wid;
    if (row >= rows) return;
    const size_t base = (size_t)row * D;
    float s = 0.f, q = 0.f;
    for (int c = lane; c < D; c += 32) { float v = x[base + c]; s += v; q += v * v; }
    #pragma unroll
    for (int o = 16; o > 0; o >>= 1) { s += __shfl_xor(s, o, 32); q += __shfl_xor(q, o, 32); }
    const float m   = s / D;
    const float var = q / D - m * m;
    const float inv = rsqrtf(var + 1e-5f);
    for (int c = lane; c < D; c += 32)
        y[base + c] = (x[base + c] - m) * inv * sc[c] + bi[c];
}

// ---------------------------------------------------------------------------
// Causal depthwise conv1d (DCONV=4) + bias + silu; input xi = xz[:, 0:768]
// ---------------------------------------------------------------------------
__global__ __launch_bounds__(256)
void conv_silu(const float* __restrict__ xz, const float* __restrict__ w,
               const float* __restrict__ b, float* __restrict__ out)
{
    const long i = (long)blockIdx.x * 256 + threadIdx.x;
    if (i >= (long)BLROWS * DINNER) return;
    const int  d  = (int)(i % DINNER);
    const long bl = i / DINNER;
    const int  l  = (int)(bl % LL);
    const long bbase = bl - l;                   // b*L
    float acc = b[d];
    #pragma unroll
    for (int k = 0; k < DCONVN; ++k) {
        const int li = l - (DCONVN - 1) + k;
        if (li >= 0)
            acc += w[d * DCONVN + k] * xz[(bbase + li) * (2 * DINNER) + d];
    }
    out[bl * DINNER + d] = acc / (1.0f + expf(-acc));   // silu
}

// ---------------------------------------------------------------------------
// Selective scan over L, fused with +u*D and y*silu(z). One thread per (b,d),
// DSTATE=16 states in registers. Sequential in t (true recurrence).
// ---------------------------------------------------------------------------
__global__ __launch_bounds__(256)
void scan_gate(const float* __restrict__ u,     // [BL,768]
               const float* __restrict__ dt,    // [BL,768]
               const float* __restrict__ dbl,   // [BL,56]  (B at +24, C at +40)
               const float* __restrict__ xz,    // [BL,1536] (z at +768)
               const float* __restrict__ A_log, // [768,16]
               const float* __restrict__ Dp,    // [768]
               float* __restrict__ y)           // [BL,768]
{
    const int i = blockIdx.x * 256 + threadIdx.x;
    if (i >= BB * DINNER) return;
    const int d = i % DINNER;
    const int b = i / DINNER;

    float Av[DSTATE];
    #pragma unroll
    for (int s = 0; s < DSTATE; ++s) Av[s] = -expf(A_log[d * DSTATE + s]);
    const float Dv = Dp[d];

    float h[DSTATE];
    #pragma unroll
    for (int s = 0; s < DSTATE; ++s) h[s] = 0.f;

    const size_t rbase = (size_t)b * LL;
    for (int t = 0; t < LL; ++t) {
        const size_t row = rbase + t;
        const float dtv = dt[row * DINNER + d];
        const float uv  = u [row * DINNER + d];
        const float* Bm = &dbl[row * 56 + DTRANK];
        const float* Cm = Bm + DSTATE;
        const float du  = dtv * uv;
        float acc = 0.f;
        #pragma unroll
        for (int s = 0; s < DSTATE; ++s) {
            const float dA = __expf(dtv * Av[s]);
            h[s] = h[s] * dA + du * Bm[s];
            acc += h[s] * Cm[s];
        }
        float yv = acc + uv * Dv;
        const float z = xz[row * (2 * DINNER) + DINNER + d];
        yv *= z / (1.f + expf(-z));               // * silu(z)
        y[row * DINNER + d] = yv;
    }
}

// ---------------------------------------------------------------------------
// Host-side orchestration
// ---------------------------------------------------------------------------
extern "C" void kernel_launch(void* const* d_in, const int* in_sizes, int n_in,
                              void* d_out, int out_size, void* d_ws, size_t ws_size,
                              hipStream_t stream)
{
    const float* in_x      = (const float*)d_in[0];
    const float* pos_emb   = (const float*)d_in[1];
    const float* ln1_s     = (const float*)d_in[2];
    const float* ln1_b     = (const float*)d_in[3];
    const float* in_w      = (const float*)d_in[4];
    const float* conv_w    = (const float*)d_in[5];
    const float* conv_b    = (const float*)d_in[6];
    const float* xproj_w   = (const float*)d_in[7];
    const float* dtproj_w  = (const float*)d_in[8];
    const float* dtproj_b  = (const float*)d_in[9];
    const float* A_log     = (const float*)d_in[10];
    const float* Dp        = (const float*)d_in[11];
    const float* out_w     = (const float*)d_in[12];
    const float* ff_ln_s   = (const float*)d_in[13];
    const float* ff_ln_b   = (const float*)d_in[14];
    const float* ff_w1     = (const float*)d_in[15];
    const float* ff_b1     = (const float*)d_in[16];
    const float* ff_w2     = (const float*)d_in[17];
    const float* ff_b2     = (const float*)d_in[18];
    const float* lno_s     = (const float*)d_in[19];
    const float* lno_b     = (const float*)d_in[20];
    (void)in_sizes; (void)n_in; (void)out_size; (void)ws_size;

    const int BL = BLROWS;
    float* X  = (float*)d_ws;          // [BL,384] residual stream
    float* T0 = X  + (size_t)BL * 384; // [BL,1536] xz
    float* T1 = T0 + (size_t)BL * 1536;// [BL,768]  xi (post conv+silu)
    float* T2 = T1 + (size_t)BL * 768; // [BL,768]  dt / ffn hidden
    float* T3 = T2 + (size_t)BL * 768; // [BL,56]   dbl
    float* T4 = T3 + (size_t)BL * 56;  // [BL,768]  y (scan out, gated)
    float* T5 = T4 + (size_t)BL * 768; // [BL,384]  LN scratch

    auto gemm = [&](const float* Ap, int lda, const float* Wp, int ldw,
                    int K, int N, const float* bias, float* Cp, int ldc,
                    const float* resid, int epi) {
        dim3 grid((N + 63) / 64, BL / 128);
        if ((K % 64 == 0) && (N % 64 == 0))
            gemm_wmma<true><<<grid, 256, 0, stream>>>(Ap, lda, Wp, ldw, K, N, bias, Cp, ldc, resid, epi);
        else
            gemm_wmma<false><<<grid, 256, 0, stream>>>(Ap, lda, Wp, ldw, K, N, bias, Cp, ldc, resid, epi);
    };

    // x = x + pos_emb
    {
        const int n = BL * DIMM;
        add_pos<<<(n + 255) / 256, 256, 0, stream>>>(in_x, pos_emb, X, n, LL * DIMM);
    }

    for (int i = 0; i < DEPTHN; ++i) {
        // ---- Mamba block ----
        ln_rows<<<BL / 8, 256, 0, stream>>>(X, ln1_s + i * DIMM, ln1_b + i * DIMM, T5, BL, DIMM);

        // xz = ln(x) @ in_w^T   [BL,384]x[1536,384]^T   (aligned)
        gemm(T5, DIMM, in_w + (size_t)i * 2 * DINNER * DIMM, DIMM,
             DIMM, 2 * DINNER, nullptr, T0, 2 * DINNER, nullptr, EPI_NONE);

        // xi = silu(conv(xi) + b)
        {
            const long n = (long)BLROWS * DINNER;
            conv_silu<<<(unsigned)((n + 255) / 256), 256, 0, stream>>>(
                T0, conv_w + (size_t)i * DINNER * DCONVN, conv_b + (size_t)i * DINNER, T1);
        }

        // dbl = xi @ xproj_w^T  [BL,768]x[56,768]^T     (N=56 -> guarded)
        gemm(T1, DINNER, xproj_w + (size_t)i * 56 * DINNER, DINNER,
             DINNER, 56, nullptr, T3, 56, nullptr, EPI_NONE);

        // dt = softplus(dbl[:, :24] @ dtproj_w^T + dtproj_b)   (K=24 -> guarded)
        gemm(T3, 56, dtproj_w + (size_t)i * DINNER * DTRANK, DTRANK,
             DTRANK, DINNER, dtproj_b + (size_t)i * DINNER, T2, DINNER, nullptr, EPI_BIAS_SOFTPLUS);

        // selective scan + u*D + gate with silu(z)
        scan_gate<<<(BB * DINNER) / 256, 256, 0, stream>>>(
            T1, T2, T3, T0,
            A_log + (size_t)i * DINNER * DSTATE, Dp + (size_t)i * DINNER, T4);

        // x += y @ out_w^T  [BL,768]x[384,768]^T  (residual fused, aligned)
        gemm(T4, DINNER, out_w + (size_t)i * DIMM * DINNER, DINNER,
             DINNER, DIMM, nullptr, X, DIMM, X, EPI_RESADD);

        // ---- FFN block ----
        ln_rows<<<BL / 8, 256, 0, stream>>>(X, ff_ln_s + i * DIMM, ff_ln_b + i * DIMM, T5, BL, DIMM);

        // h = gelu(ln(x) @ ff_w1^T + b1)   (aligned)
        gemm(T5, DIMM, ff_w1 + (size_t)i * DINNER * DIMM, DIMM,
             DIMM, DINNER, ff_b1 + (size_t)i * DINNER, T2, DINNER, nullptr, EPI_BIAS_GELU);

        // x += h @ ff_w2^T + b2  (bias + residual fused, aligned)
        gemm(T2, DINNER, ff_w2 + (size_t)i * DIMM * DINNER, DINNER,
             DINNER, DIMM, ff_b2 + (size_t)i * DIMM, X, DIMM, X, EPI_BIAS_RESADD);
    }

    // final layer norm -> d_out
    ln_rows<<<BL / 8, 256, 0, stream>>>(X, lno_s, lno_b, (float*)d_out, BL, DIMM);
}
